// SharedGATEncoder_32006096290083
// MI455X (gfx1250) — compile-verified, matrix operand order checked by reading, not run on hardware
//
#include <hip/hip_runtime.h>
#include <hip/hip_bf16.h>
#include <math.h>

// ---------------------------------------------------------------------------
// SharedGATEncoder for MI455X (gfx1250, wave32, WMMA).
//   prep : f32->f16 weight transposes, fold Wq/Wk with att vectors -> [4,128]
//   gemm : v_wmma_f32_16x16x32_f16, 2 col-tiles / iter; all 8 B fragments
//          loaded up front per iteration (one big load clause, staggered
//          s_wait_loadcnt) feeding two independent accumulator chains
//   edges: alpha = aq[dst]+ak[src]; atomic segment max; exp; atomic segment
//          sum; weighted scatter of v with f32 atomics
//   node : residual + LayerNorm (+ReLU) per wave32 with shfl_xor reductions
// ---------------------------------------------------------------------------

#define DIM   128
#define HEADS 4
#define C1HD  32
#define C2HD  128
#define LNEPS 1e-5f

typedef _Float16 h8   __attribute__((ext_vector_type(8)));
typedef _Float16 v16h __attribute__((ext_vector_type(16)));
typedef float    v8f  __attribute__((ext_vector_type(8)));

// ---------------- WMMA fragment helpers -----------------------------------
// 16-bit A (16x32) layout: lanes 0-15 hold K{0..7,16..23}, lanes 16-31 hold
// K{8..15,24..31} of row M=lane&15.  B (32x16) mirrored with N=lane&15.
__device__ __forceinline__ v16h load_frag16(const _Float16* base, int koff) {
  h8 lo = *(const h8*)(base + koff);
  h8 hi = *(const h8*)(base + 16 + koff);
  return __builtin_shufflevector(lo, hi, 0,1,2,3,4,5,6,7,8,9,10,11,12,13,14,15);
}

#define WMMA_F16(AF, BF, CC) \
  __builtin_amdgcn_wmma_f32_16x16x32_f16(false, (AF), false, (BF), (short)0, (CC), false, false)

// out = act(A @ WT^T + bias).  A: [M x 128] f16 rowmajor.  WT: [NC x 128] f16
// (pre-transposed weights).  4 waves/block, 16 rows per wave, full K=128,
// two 16-col tiles per iteration -> two independent WMMA chains.
template<int RELU, int HAS16>
__global__ void __launch_bounds__(128) gemm_wmma_f16(
    const _Float16* __restrict__ A, const _Float16* __restrict__ WT,
    const float* __restrict__ bias, float* __restrict__ outF,
    _Float16* __restrict__ outH, int M, int NC)
{
  const int lane  = threadIdx.x & 31;
  const int wave  = threadIdx.x >> 5;
  const int mbase = blockIdx.x * 64 + wave * 16;
  if (mbase >= M) return;                 // wave-uniform: EXEC stays all-1s
  const int l15  = lane & 15;
  const int koff = (lane >> 4) << 3;
  const int r0   = (lane >> 4) << 3;

  const _Float16* arow = A + (size_t)(mbase + l15) * DIM;
  const v16h a0 = load_frag16(arow +  0, koff);
  const v16h a1 = load_frag16(arow + 32, koff);
  const v16h a2 = load_frag16(arow + 64, koff);
  const v16h a3 = load_frag16(arow + 96, koff);

  for (int nt = 0; nt < NC; nt += 32) {
    const int colX = nt + l15;
    const int colY = colX + 16;
    const _Float16* bx = WT + (size_t)colX * DIM;
    const _Float16* by = WT + (size_t)colY * DIM;

    // Issue all 16 b128 loads first: one clause, staggered waits under WMMAs.
    const v16h bx0 = load_frag16(bx +  0, koff);
    const v16h bx1 = load_frag16(bx + 32, koff);
    const v16h bx2 = load_frag16(bx + 64, koff);
    const v16h bx3 = load_frag16(bx + 96, koff);
    const v16h by0 = load_frag16(by +  0, koff);
    const v16h by1 = load_frag16(by + 32, koff);
    const v16h by2 = load_frag16(by + 64, koff);
    const v16h by3 = load_frag16(by + 96, koff);

    v8f accX = {}, accY = {};
    accX = WMMA_F16(a0, bx0, accX);
    accY = WMMA_F16(a0, by0, accY);
    accX = WMMA_F16(a1, bx1, accX);
    accY = WMMA_F16(a1, by1, accY);
    accX = WMMA_F16(a2, bx2, accX);
    accY = WMMA_F16(a2, by2, accY);
    accX = WMMA_F16(a3, bx3, accX);
    accY = WMMA_F16(a3, by3, accY);

    const float bvX = bias[colX];
    const float bvY = bias[colY];
    #pragma unroll
    for (int r = 0; r < 8; ++r) {         // C/D: VGPR r -> M = r0 + r
      const size_t orow = (size_t)(mbase + r0 + r) * NC;
      float vx = accX[r] + bvX;
      float vy = accY[r] + bvY;
      if (RELU) { vx = fmaxf(vx, 0.f); vy = fmaxf(vy, 0.f); }
      outF[orow + colX] = vx;
      outF[orow + colY] = vy;
      if (HAS16) {
        outH[orow + colX] = (_Float16)vx;
        outH[orow + colY] = (_Float16)vy;
      }
    }
  }
}

// ---------------- prep kernels ---------------------------------------------
__global__ void cvt_f32_to_f16(const float* __restrict__ s, _Float16* __restrict__ d, int n) {
  int t = blockIdx.x * blockDim.x + threadIdx.x;
  if (t < n) d[t] = (_Float16)s[t];
}

// WT[c*128 + r] = f16(W[r*cols + c])   (W is [128 x cols])
__global__ void transpose_f16(const float* __restrict__ W, _Float16* __restrict__ WT, int cols) {
  int t = blockIdx.x * blockDim.x + threadIdx.x;
  if (t >= cols * DIM) return;
  int c = t / DIM, r = t - c * DIM;
  WT[t] = (_Float16)W[(size_t)r * cols + c];
}

// Fold Wq/Wk [128 x H*C] with attention half-vector -> Weff [4 x 128], beff[4]
__global__ void fold_att(const float* __restrict__ W, const float* __restrict__ b,
                         const float* __restrict__ att, int C, int useK,
                         float* __restrict__ Weff, float* __restrict__ beff)
{
  int t = blockIdx.x * blockDim.x + threadIdx.x;
  if (t >= DIM * HEADS) return;
  int d = t >> 2, hh = t & 3;
  const float* av   = att + hh * 2 * C + (useK ? C : 0);
  const float* wrow = W + (size_t)d * (HEADS * C) + hh * C;
  float acc = 0.f;
  for (int c = 0; c < C; ++c) acc += wrow[c] * av[c];
  Weff[hh * DIM + d] = acc;               // [head][d] for contiguous reads
  if (d == 0) {
    const float* brow = b + hh * C;
    float bb = 0.f;
    for (int c = 0; c < C; ++c) bb += brow[c] * av[c];
    beff[hh] = bb;
  }
}

// aq/ak[n,h] = h[n,:] @ Weff[h,:] + beff[h]   (tiny [N,128]@[128,4])
__global__ void node_att(const float* __restrict__ hmat, const float* __restrict__ Weff,
                         const float* __restrict__ beff, float* __restrict__ out, int N)
{
  int t = blockIdx.x * blockDim.x + threadIdx.x;
  if (t >= N * HEADS) return;
  int n = t >> 2, hh = t & 3;
  const float* row = hmat + (size_t)n * DIM;
  const float* w   = Weff + hh * DIM;
  float acc = beff[hh];
  for (int d = 0; d < DIM; ++d) acc += row[d] * w[d];
  out[t] = acc;
}

__global__ void fill_f32(float* __restrict__ p, float v, int n) {
  int t = blockIdx.x * blockDim.x + threadIdx.x;
  if (t < n) p[t] = v;
}

// ---------------- edge kernels ---------------------------------------------
__device__ __forceinline__ void atomicMaxF(float* a, float v) {
  // sign-split int trick -> global_atomic_max_i32 / min_u32 (always available)
  if (v >= 0.f) atomicMax((int*)a, __float_as_int(v));
  else          atomicMin((unsigned int*)a, __float_as_uint(v));
}

__global__ void edge_alpha_max(const int* __restrict__ src, const int* __restrict__ dst,
                               const float* __restrict__ aq, const float* __restrict__ ak,
                               float* __restrict__ alpha, float* __restrict__ m, int E)
{
  int t = blockIdx.x * blockDim.x + threadIdx.x;
  if (t >= E * HEADS) return;
  int e = t >> 2, hh = t & 3;
  int si = src[e], di = dst[e];
  float a = aq[di * HEADS + hh] + ak[si * HEADS + hh];
  alpha[t] = a;
  atomicMaxF(&m[di * HEADS + hh], a);
}

__global__ void edge_exp_sum(const int* __restrict__ dst, float* __restrict__ alpha,
                             const float* __restrict__ m, float* __restrict__ s, int E)
{
  int t = blockIdx.x * blockDim.x + threadIdx.x;
  if (t >= E * HEADS) return;
  int e = t >> 2, hh = t & 3;
  int di = dst[e];
  float ex = __expf(alpha[t] - m[di * HEADS + hh]);
  alpha[t] = ex;                          // overwrite in place
  atomicAdd(&s[di * HEADS + hh], ex);
}

// layer1: v [N, H*32], concat heads.  One wave per edge: lane -> (head, c4).
__global__ void __launch_bounds__(256) scatter_c32(
    const int* __restrict__ src, const int* __restrict__ dst,
    const float* __restrict__ ex, const float* __restrict__ ssum,
    const float* __restrict__ v, float* __restrict__ o, int E)
{
  int e = blockIdx.x * 8 + (threadIdx.x >> 5);
  if (e >= E) return;
  int lane = threadIdx.x & 31;
  int hh = lane >> 3, c0 = (lane & 7) << 2;
  int si = src[e], di = dst[e];
  float a = ex[e * HEADS + hh] / fmaxf(ssum[di * HEADS + hh], 1e-16f);
  float4 vv = *(const float4*)(v + (size_t)si * DIM + hh * C1HD + c0);
  float* op = o + (size_t)di * DIM + hh * C1HD + c0;
  atomicAdd(op + 0, vv.x * a); atomicAdd(op + 1, vv.y * a);
  atomicAdd(op + 2, vv.z * a); atomicAdd(op + 3, vv.w * a);
}

// layer2: v [N, H*128], mean over heads.  One wave per edge: lane -> 4 cols.
__global__ void __launch_bounds__(256) scatter_c128(
    const int* __restrict__ src, const int* __restrict__ dst,
    const float* __restrict__ ex, const float* __restrict__ ssum,
    const float* __restrict__ v, float* __restrict__ o, int E)
{
  int e = blockIdx.x * 8 + (threadIdx.x >> 5);
  if (e >= E) return;
  int lane = threadIdx.x & 31;
  int c0 = lane << 2;
  int si = src[e], di = dst[e];
  float ax = 0.f, ay = 0.f, az = 0.f, aw = 0.f;
  #pragma unroll
  for (int hh = 0; hh < HEADS; ++hh) {
    float a = ex[e * HEADS + hh] / fmaxf(ssum[di * HEADS + hh], 1e-16f);
    float4 vv = *(const float4*)(v + (size_t)si * (HEADS * C2HD) + hh * C2HD + c0);
    ax += vv.x * a; ay += vv.y * a; az += vv.z * a; aw += vv.w * a;
  }
  const float inv = 1.f / HEADS;
  float* op = o + (size_t)di * DIM + c0;
  atomicAdd(op + 0, ax * inv); atomicAdd(op + 1, ay * inv);
  atomicAdd(op + 2, az * inv); atomicAdd(op + 3, aw * inv);
}

// ---------------- residual + LayerNorm (+ReLU), one wave32 per node --------
template<int RELU>
__global__ void __launch_bounds__(256) resid_ln(
    const float* __restrict__ o, const float* __restrict__ res,
    const float* __restrict__ g, const float* __restrict__ beta,
    float* __restrict__ outF, _Float16* __restrict__ outH, int N)
{
  int n = blockIdx.x * 8 + (threadIdx.x >> 5);
  if (n >= N) return;
  int lane = threadIdx.x & 31;
  const int c0 = lane << 2;
  float4 a = *(const float4*)(o   + (size_t)n * DIM + c0);
  float4 b = *(const float4*)(res + (size_t)n * DIM + c0);
  float x0 = a.x + b.x, x1 = a.y + b.y, x2 = a.z + b.z, x3 = a.w + b.w;
  float s1 = x0 + x1 + x2 + x3;
  #pragma unroll
  for (int off = 16; off > 0; off >>= 1) s1 += __shfl_xor(s1, off, 32);
  const float mu = s1 * (1.f / DIM);
  float d0 = x0 - mu, d1 = x1 - mu, d2 = x2 - mu, d3 = x3 - mu;
  float s2 = d0 * d0 + d1 * d1 + d2 * d2 + d3 * d3;
  #pragma unroll
  for (int off = 16; off > 0; off >>= 1) s2 += __shfl_xor(s2, off, 32);
  const float inv = rsqrtf(s2 * (1.f / DIM) + LNEPS);
  float y[4] = {d0, d1, d2, d3};
  #pragma unroll
  for (int i = 0; i < 4; ++i) {
    int c = c0 + i;
    float v = y[i] * inv * g[c] + beta[c];
    if (RELU) v = fmaxf(v, 0.f);
    outF[(size_t)n * DIM + c] = v;
    outH[(size_t)n * DIM + c] = (_Float16)v;
  }
}

// ---------------------------------------------------------------------------
extern "C" void kernel_launch(void* const* d_in, const int* in_sizes, int n_in,
                              void* d_out, int out_size, void* d_ws, size_t ws_size,
                              hipStream_t stream) {
  const float* x    = (const float*)d_in[0];
  const int*   ei   = (const int*)  d_in[1];
  const float* W_in = (const float*)d_in[2];  const float* b_in = (const float*)d_in[3];
  const float* Wq1  = (const float*)d_in[4];  const float* bq1  = (const float*)d_in[5];
  const float* Wk1  = (const float*)d_in[6];  const float* bk1  = (const float*)d_in[7];
  const float* Wv1  = (const float*)d_in[8];  const float* bv1  = (const float*)d_in[9];
  const float* att1 = (const float*)d_in[10];
  const float* g1   = (const float*)d_in[11]; const float* be1  = (const float*)d_in[12];
  const float* Wq2  = (const float*)d_in[13]; const float* bq2  = (const float*)d_in[14];
  const float* Wk2  = (const float*)d_in[15]; const float* bk2  = (const float*)d_in[16];
  const float* Wv2  = (const float*)d_in[17]; const float* bv2  = (const float*)d_in[18];
  const float* att2 = (const float*)d_in[19];
  const float* g2   = (const float*)d_in[20]; const float* be2  = (const float*)d_in[21];
  const float* Wout = (const float*)d_in[22]; const float* bout = (const float*)d_in[23];

  const int N = in_sizes[0] / DIM;
  const int E = in_sizes[1] / 2;
  const int* srcI = ei;
  const int* dstI = ei + E;

  // bump allocator over workspace (256B aligned regions)
  char* wp = (char*)d_ws;
  auto alloc = [&](size_t bytes) -> void* {
    void* r = wp; wp += (bytes + 255) & ~(size_t)255; return r;
  };
  _Float16* x16   = (_Float16*)alloc((size_t)N * DIM * 2);
  _Float16* h16   = (_Float16*)alloc((size_t)N * DIM * 2);
  _Float16* WTin  = (_Float16*)alloc((size_t)DIM * DIM * 2);
  _Float16* WTv1  = (_Float16*)alloc((size_t)DIM * DIM * 2);
  _Float16* WTv2  = (_Float16*)alloc((size_t)(HEADS * C2HD) * DIM * 2);
  _Float16* WTout = (_Float16*)alloc((size_t)DIM * DIM * 2);
  float* hA   = (float*)alloc((size_t)N * DIM * 4);
  float* hB   = (float*)alloc((size_t)N * DIM * 4);
  float* vbuf = (float*)alloc((size_t)N * (HEADS * C2HD) * 4);   // v1 uses first N*128
  float* obuf = (float*)alloc((size_t)N * DIM * 4);
  float* aq   = (float*)alloc((size_t)N * HEADS * 4);
  float* ak   = (float*)alloc((size_t)N * HEADS * 4);
  float* mBuf = (float*)alloc((size_t)N * HEADS * 4);
  float* sBuf = (float*)alloc((size_t)N * HEADS * 4);
  float* exB  = (float*)alloc((size_t)E * HEADS * 4);
  float* WeQ1 = (float*)alloc(DIM * HEADS * 4); float* beQ1 = (float*)alloc(HEADS * 4);
  float* WeK1 = (float*)alloc(DIM * HEADS * 4); float* beK1 = (float*)alloc(HEADS * 4);
  float* WeQ2 = (float*)alloc(DIM * HEADS * 4); float* beQ2 = (float*)alloc(HEADS * 4);
  float* WeK2 = (float*)alloc(DIM * HEADS * 4); float* beK2 = (float*)alloc(HEADS * 4);

  auto cdiv = [](int a, int b) { return (a + b - 1) / b; };
  const int NH = N * HEADS, EH = E * HEADS;

  // --- prep: weight transposes (f16), attention folds, x -> f16 ----------
  transpose_f16<<<cdiv(DIM * DIM, 256), 256, 0, stream>>>(W_in, WTin, DIM);
  transpose_f16<<<cdiv(DIM * DIM, 256), 256, 0, stream>>>(Wv1, WTv1, DIM);
  transpose_f16<<<cdiv(HEADS * C2HD * DIM, 256), 256, 0, stream>>>(Wv2, WTv2, HEADS * C2HD);
  transpose_f16<<<cdiv(DIM * DIM, 256), 256, 0, stream>>>(Wout, WTout, DIM);
  fold_att<<<cdiv(DIM * HEADS, 256), 256, 0, stream>>>(Wq1, bq1, att1, C1HD, 0, WeQ1, beQ1);
  fold_att<<<cdiv(DIM * HEADS, 256), 256, 0, stream>>>(Wk1, bk1, att1, C1HD, 1, WeK1, beK1);
  fold_att<<<cdiv(DIM * HEADS, 256), 256, 0, stream>>>(Wq2, bq2, att2, C2HD, 0, WeQ2, beQ2);
  fold_att<<<cdiv(DIM * HEADS, 256), 256, 0, stream>>>(Wk2, bk2, att2, C2HD, 1, WeK2, beK2);
  cvt_f32_to_f16<<<cdiv(N * DIM, 256), 256, 0, stream>>>(x, x16, N * DIM);

  // --- h0 = relu(x @ W_in + b_in) ----------------------------------------
  gemm_wmma_f16<1, 1><<<cdiv(N, 64), 128, 0, stream>>>(x16, WTin, b_in, hA, h16, N, DIM);

  // --- GAT layer 1 --------------------------------------------------------
  gemm_wmma_f16<0, 0><<<cdiv(N, 64), 128, 0, stream>>>(h16, WTv1, bv1, vbuf, nullptr, N, DIM);
  node_att<<<cdiv(NH, 256), 256, 0, stream>>>(hA, WeQ1, beQ1, aq, N);
  node_att<<<cdiv(NH, 256), 256, 0, stream>>>(hA, WeK1, beK1, ak, N);
  fill_f32<<<cdiv(NH, 256), 256, 0, stream>>>(mBuf, -1e30f, NH);
  fill_f32<<<cdiv(NH, 256), 256, 0, stream>>>(sBuf, 0.f, NH);
  fill_f32<<<cdiv(N * DIM, 256), 256, 0, stream>>>(obuf, 0.f, N * DIM);
  edge_alpha_max<<<cdiv(EH, 256), 256, 0, stream>>>(srcI, dstI, aq, ak, exB, mBuf, E);
  edge_exp_sum<<<cdiv(EH, 256), 256, 0, stream>>>(dstI, exB, mBuf, sBuf, E);
  scatter_c32<<<cdiv(E, 8), 256, 0, stream>>>(srcI, dstI, exB, sBuf, vbuf, obuf, E);
  resid_ln<1><<<cdiv(N, 8), 256, 0, stream>>>(obuf, hA, g1, be1, hB, h16, N);

  // --- GAT layer 2 (mean over heads) -------------------------------------
  gemm_wmma_f16<0, 0><<<cdiv(N, 64), 128, 0, stream>>>(h16, WTv2, bv2, vbuf, nullptr, N, HEADS * C2HD);
  node_att<<<cdiv(NH, 256), 256, 0, stream>>>(hB, WeQ2, beQ2, aq, N);
  node_att<<<cdiv(NH, 256), 256, 0, stream>>>(hB, WeK2, beK2, ak, N);
  fill_f32<<<cdiv(NH, 256), 256, 0, stream>>>(mBuf, -1e30f, NH);
  fill_f32<<<cdiv(NH, 256), 256, 0, stream>>>(sBuf, 0.f, NH);
  fill_f32<<<cdiv(N * DIM, 256), 256, 0, stream>>>(obuf, 0.f, N * DIM);
  edge_alpha_max<<<cdiv(EH, 256), 256, 0, stream>>>(srcI, dstI, aq, ak, exB, mBuf, E);
  edge_exp_sum<<<cdiv(EH, 256), 256, 0, stream>>>(dstI, exB, mBuf, sBuf, E);
  scatter_c128<<<cdiv(E, 8), 256, 0, stream>>>(srcI, dstI, exB, sBuf, vbuf, obuf, E);
  resid_ln<0><<<cdiv(N, 8), 256, 0, stream>>>(obuf, hB, g2, be2, hA, h16, N);

  // --- z = h @ W_out + b_out ---------------------------------------------
  gemm_wmma_f16<0, 0><<<cdiv(N, 64), 128, 0, stream>>>(h16, WTout, bout, (float*)d_out, nullptr, N, DIM);
}
